// Int8Attention_23596550324651
// MI455X (gfx1250) — 8-bit matrix operand order repaired and checked by reading, compiled, not run on hardware
//
#include <hip/hip_runtime.h>

typedef signed char i8;
typedef __attribute__((ext_vector_type(8))) int v8i;
typedef __attribute__((ext_vector_type(4))) int v4i;

#define BATCH   64
#define HS      14
#define WSZ     14
#define EDIM    768
#define NHEADS  12
#define HDIM    64
#define NTOK    196
#define MROWS   12544
#define NPAD    208
#define KPAD    256

#define C_A_QKV  0.012f
#define C_B_QKV  0.8f
#define C_A_QK   0.0035f
#define C_A_PV   (1.0f/127.0f)
#define C_A_PROJ 0.002f

#define AS1 __attribute__((address_space(1)))
#define AS3 __attribute__((address_space(3)))

#if __has_builtin(__builtin_amdgcn_global_load_async_to_lds_b128)
#define USE_ASYNC_LDS 1
#else
#define USE_ASYNC_LDS 0
#endif

__device__ __forceinline__ void async_b128(i8* lds, const i8* g) {
#if USE_ASYNC_LDS
    __builtin_amdgcn_global_load_async_to_lds_b128(
        (AS1 v4i*)g, (AS3 v4i*)lds, 0, 0);
#else
    (void)lds; (void)g;
#endif
}
__device__ __forceinline__ void wait_async() {
#if USE_ASYNC_LDS
#if __has_builtin(__builtin_amdgcn_s_wait_asynccnt)
    __builtin_amdgcn_s_wait_asynccnt(0);
#else
    asm volatile("s_wait_asynccnt 0x0" ::: "memory");
#endif
#endif
}

__device__ __forceinline__ int clampi(int x, int lo, int hi) {
    return x < lo ? lo : (x > hi ? hi : x);
}

__device__ __forceinline__ v8i load_a8(const i8* base, int stride, int lane, int k0) {
    const i8* p = base + (lane & 15) * stride + k0 + ((lane >> 4) << 3);
    union { v8i v; int2 d[4]; } u;
    u.d[0] = *(const int2*)(p);
    u.d[1] = *(const int2*)(p + 16);
    u.d[2] = *(const int2*)(p + 32);
    u.d[3] = *(const int2*)(p + 48);
    return u.v;
}
__device__ __forceinline__ v8i load_b8(const i8* base, int stride, int lane, int k0) {
    const i8* p = base + (lane & 15) * stride + k0 + ((lane >> 4) << 3);
    union { v8i v; int2 d[4]; } u;
    u.d[0] = *(const int2*)(p);
    u.d[1] = *(const int2*)(p + 16);
    u.d[2] = *(const int2*)(p + 32);
    u.d[3] = *(const int2*)(p + 48);
    return u.v;
}

__device__ __forceinline__ void gemm_mainloop(const i8* __restrict__ X,
                                              const i8* __restrict__ W,
                                              i8* xs, i8* wt,
                                              int m0, int n0, int t, v8i acc[4]) {
    const int lane = t & 31, wv = t >> 5;
    const int wm = (wv & 3) * 16;
    const int wn = (wv >> 2) * 64;
    const int row = t >> 2, off = (t & 3) * 16;
    const i8* xp  = X + (size_t)(m0 + row) * EDIM + off;
    const i8* wp0 = W + (size_t)(n0 + row) * EDIM + off;
    const i8* wp1 = W + (size_t)(n0 + row + 64) * EDIM + off;

#if USE_ASYNC_LDS
    async_b128(xs + row * 64 + off, xp);
    async_b128(wt + row * 64 + off, wp0);
    async_b128(wt + (row + 64) * 64 + off, wp1);
    for (int kc = 0; kc < 12; ++kc) {
        const int cur = kc & 1, nxt = cur ^ 1;
        wait_async();
        __syncthreads();
        if (kc < 11) {
            const int k1 = (kc + 1) * 64;
            async_b128(xs + nxt * 4096 + row * 64 + off, xp + k1);
            async_b128(wt + nxt * 8192 + row * 64 + off, wp0 + k1);
            async_b128(wt + nxt * 8192 + (row + 64) * 64 + off, wp1 + k1);
        }
        i8* xb = xs + cur * 4096;
        i8* wb = wt + cur * 8192;
        v8i a = load_a8(xb + wm * 64, 64, lane, 0);
        #pragma unroll
        for (int j = 0; j < 4; ++j) {
            v8i b = load_b8(wb + (wn + j * 16) * 64, 64, lane, 0);
            acc[j] = __builtin_amdgcn_wmma_i32_16x16x64_iu8(
                true, a, true, b, acc[j], false, false);
        }
    }
#else
    int4 r0 = *(const int4*)xp;
    int4 r1 = *(const int4*)wp0;
    int4 r2 = *(const int4*)wp1;
    for (int kc = 0; kc < 12; ++kc) {
        __syncthreads();
        *(int4*)(xs + row * 64 + off) = r0;
        *(int4*)(wt + row * 64 + off) = r1;
        *(int4*)(wt + (row + 64) * 64 + off) = r2;
        if (kc < 11) {
            const int k1 = (kc + 1) * 64;
            r0 = *(const int4*)(xp + k1);
            r1 = *(const int4*)(wp0 + k1);
            r2 = *(const int4*)(wp1 + k1);
        }
        __syncthreads();
        v8i a = load_a8(xs + wm * 64, 64, lane, 0);
        #pragma unroll
        for (int j = 0; j < 4; ++j) {
            v8i b = load_b8(wt + (wn + j * 16) * 64, 64, lane, 0);
            acc[j] = __builtin_amdgcn_wmma_i32_16x16x64_iu8(
                true, a, true, b, acc[j], false, false);
        }
    }
#endif
}

__global__ __launch_bounds__(256) void pack_i8(const int* __restrict__ s,
                                               i8* __restrict__ d, int n4) {
    int i = blockIdx.x * 256 + threadIdx.x;
    if (i < n4) {
        int4 v = ((const int4*)s)[i];
        ((int*)d)[i] = (v.x & 255) | ((v.y & 255) << 8) |
                       ((v.z & 255) << 16) | (v.w << 24);
    }
}

__global__ __launch_bounds__(256) void qkv_gemm(const i8* __restrict__ X,
                                                const i8* __restrict__ W,
                                                const int* __restrict__ bias,
                                                i8* __restrict__ Oh) {
    __shared__ __align__(16) i8 xs[2 * 64 * 64];
    __shared__ __align__(16) i8 wt[2 * 128 * 64];
    const int m0 = blockIdx.x * 64;
    const int n0 = blockIdx.y * 128;
    const int t = threadIdx.x, lane = t & 31, wv = t >> 5;
    const int wm = (wv & 3) * 16, wn = (wv >> 2) * 64;

    v8i acc[4] = {v8i{}, v8i{}, v8i{}, v8i{}};
    gemm_mainloop(X, W, xs, wt, m0, n0, t, acc);

    const int col16 = lane & 15, hh = lane >> 4;
    #pragma unroll
    for (int j = 0; j < 4; ++j) {
        int c = n0 + wn + j * 16 + col16;
        float bv = C_B_QKV * (float)bias[c];
        int nh = c >> 6, hd = c & 63;
        #pragma unroll
        for (int r = 0; r < 8; ++r) {
            int m = m0 + wm + r + 8 * hh;
            int y = clampi((int)rintf(C_A_QKV * (float)acc[j][r] + bv), -128, 127);
            int bb = m / NTOK, np = m % NTOK;
            Oh[(((size_t)(bb * NHEADS + nh) * NTOK) + np) * HDIM + hd] = (i8)y;
        }
    }
}

#define SH_Q   0
#define SH_K   (NPAD * HDIM)
#define SH_VT  (2 * NPAD * HDIM)
#define SH_RH  (SH_VT + HDIM * KPAD)
#define SH_RW  (SH_RH + 2744 * 4)
#define SH_SC  (SH_RW + 2744 * 4)
#define SH_PR  (SH_SC + NPAD * NPAD * 4)
#define SH_TOT (SH_PR + NPAD * KPAD)

__global__ __launch_bounds__(256) void attn_head(const i8* __restrict__ Qh,
                                                 const i8* __restrict__ Kh,
                                                 const i8* __restrict__ Vh,
                                                 const float* __restrict__ rph,
                                                 const float* __restrict__ rpw,
                                                 i8* __restrict__ O8) {
    extern __shared__ __align__(16) char sm[];
    i8*    q8s  = (i8*)(sm + SH_Q);
    i8*    k8s  = (i8*)(sm + SH_K);
    i8*    vTs  = (i8*)(sm + SH_VT);
    float* relh = (float*)(sm + SH_RH);
    float* relw = (float*)(sm + SH_RW);
    float* sc   = (float*)(sm + SH_SC);
    i8*    pr   = (i8*)(sm + SH_PR);

    const int head = blockIdx.x;
    const int t = threadIdx.x, lane = t & 31, wv = t >> 5;
    const i8* qg = Qh + (size_t)head * NTOK * HDIM;
    const i8* kg = Kh + (size_t)head * NTOK * HDIM;
    const i8* vg = Vh + (size_t)head * NTOK * HDIM;

#if USE_ASYNC_LDS
    for (int i = t; i < (NTOK * HDIM) / 16; i += 256) {
        async_b128(q8s + i * 16, qg + i * 16);
        async_b128(k8s + i * 16, kg + i * 16);
    }
#else
    for (int i = t; i < (NTOK * HDIM) / 16; i += 256) {
        ((int4*)q8s)[i] = ((const int4*)qg)[i];
        ((int4*)k8s)[i] = ((const int4*)kg)[i];
    }
#endif
    const int4 z4 = make_int4(0, 0, 0, 0);
    for (int i = t; i < (HDIM * KPAD) / 16; i += 256) ((int4*)vTs)[i] = z4;
    for (int i = t; i < (NPAD * KPAD) / 16; i += 256) ((int4*)pr)[i] = z4;
    for (int i = t; i < ((NPAD - NTOK) * HDIM) / 16; i += 256) {
        ((int4*)(q8s + NTOK * HDIM))[i] = z4;
        ((int4*)(k8s + NTOK * HDIM))[i] = z4;
    }
    wait_async();
    __syncthreads();
    for (int i = t; i < NTOK * HDIM; i += 256) {
        int k = i >> 6, n = i & 63;
        vTs[n * KPAD + k] = vg[i];
    }
    for (int i = t; i < 2744; i += 256) {
        int k = i % 14, w = (i / 14) % 14, h = i / 196;
        const i8* qrow = q8s + (h * 14 + w) * HDIM;
        const float* rh = rph + (h - k + 13) * HDIM;
        const float* rw = rpw + (w - k + 13) * HDIM;
        float sh = 0.f, sw = 0.f;
        for (int c = 0; c < HDIM; ++c) {
            float qf = (float)qrow[c];
            sh += qf * rh[c];
            sw += qf * rw[c];
        }
        relh[i] = 0.01f * sh;
        relw[i] = 0.01f * sw;
    }
    __syncthreads();

    for (int tile = wv; tile < 169; tile += 8) {
        int Mi = tile / 13, Ni = tile % 13;
        v8i a = load_a8(q8s + Mi * 16 * HDIM, HDIM, lane, 0);
        v8i b = load_b8(k8s + Ni * 16 * HDIM, HDIM, lane, 0);
        v8i c = v8i{};
        c = __builtin_amdgcn_wmma_i32_16x16x64_iu8(true, a, true, b, c, false, false);
        int n = Ni * 16 + (lane & 15);
        #pragma unroll
        for (int r = 0; r < 8; ++r) {
            int m = Mi * 16 + r + 8 * (lane >> 4);
            float s;
            if (n >= NTOK) {
                s = -1e30f;
            } else {
                int hw = (m < NTOK) ? m : 0;
                int hp = n / 14, wp = n % 14;
                s = C_A_QK * (float)c[r] + relh[hw * 14 + hp] + relw[hw * 14 + wp];
            }
            sc[m * NPAD + n] = s;
        }
    }
    __syncthreads();

    for (int row = wv; row < NTOK; row += 8) {
        const float* r = sc + row * NPAD;
        float mx = -1e30f;
        for (int j = lane; j < NPAD; j += 32) mx = fmaxf(mx, r[j]);
        for (int off = 16; off > 0; off >>= 1) mx = fmaxf(mx, __shfl_xor(mx, off, 32));
        float sum = 0.f;
        for (int j = lane; j < NPAD; j += 32) sum += __expf(r[j] - mx);
        for (int off = 16; off > 0; off >>= 1) sum += __shfl_xor(sum, off, 32);
        float inv = 127.0f / sum;
        for (int j = lane; j < NPAD; j += 32) {
            int p = clampi((int)rintf(__expf(r[j] - mx) * inv), -128, 127);
            pr[row * KPAD + j] = (i8)p;
        }
    }
    __syncthreads();

    const int bb = head / NHEADS, nh = head % NHEADS;
    for (int tile = wv; tile < 52; tile += 8) {
        int Mi = tile >> 2, Ni = tile & 3;
        v8i c = v8i{};
        #pragma unroll
        for (int kc = 0; kc < 4; ++kc) {
            v8i a = load_a8(pr + Mi * 16 * KPAD, KPAD, lane, kc * 64);
            v8i b = load_b8(vTs + Ni * 16 * KPAD, KPAD, lane, kc * 64);
            c = __builtin_amdgcn_wmma_i32_16x16x64_iu8(true, a, true, b, c, false, false);
        }
        int hd = Ni * 16 + (lane & 15);
        #pragma unroll
        for (int r = 0; r < 8; ++r) {
            int m = Mi * 16 + r + 8 * (lane >> 4);
            if (m < NTOK) {
                int y = clampi((int)rintf((float)c[r] * C_A_PV), -128, 127);
                O8[((size_t)(bb * NTOK + m)) * EDIM + nh * HDIM + hd] = (i8)y;
            }
        }
    }
}

__global__ __launch_bounds__(256) void proj_gemm(const i8* __restrict__ X,
                                                 const i8* __restrict__ W,
                                                 const float* __restrict__ bias,
                                                 float* __restrict__ Out) {
    __shared__ __align__(16) i8 xs[2 * 64 * 64];
    __shared__ __align__(16) i8 wt[2 * 128 * 64];
    const int m0 = blockIdx.x * 64;
    const int n0 = blockIdx.y * 128;
    const int t = threadIdx.x, lane = t & 31, wv = t >> 5;
    const int wm = (wv & 3) * 16, wn = (wv >> 2) * 64;

    v8i acc[4] = {v8i{}, v8i{}, v8i{}, v8i{}};
    gemm_mainloop(X, W, xs, wt, m0, n0, t, acc);

    const int col16 = lane & 15, hh = lane >> 4;
    #pragma unroll
    for (int j = 0; j < 4; ++j) {
        int c = n0 + wn + j * 16 + col16;
        float bv = bias[c];
        #pragma unroll
        for (int r = 0; r < 8; ++r) {
            int m = m0 + wm + r + 8 * hh;
            Out[(size_t)m * EDIM + c] = C_A_PROJ * (float)acc[j][r] + bv;
        }
    }
}

extern "C" void kernel_launch(void* const* d_in, const int* in_sizes, int n_in,
                              void* d_out, int out_size, void* d_ws, size_t ws_size,
                              hipStream_t stream) {
    (void)in_sizes; (void)n_in; (void)out_size; (void)ws_size;

    const int*   x    = (const int*)d_in[0];
    const int*   q_w  = (const int*)d_in[1];
    const int*   q_b  = (const int*)d_in[2];
    const int*   k_w  = (const int*)d_in[3];
    const int*   k_b  = (const int*)d_in[4];
    const int*   v_w  = (const int*)d_in[5];
    const int*   v_b  = (const int*)d_in[6];
    const int*   p_w  = (const int*)d_in[7];
    const float* p_b  = (const float*)d_in[8];
    const float* rph  = (const float*)d_in[9];
    const float* rpw  = (const float*)d_in[10];
    float* out = (float*)d_out;

    const size_t XB = (size_t)MROWS * EDIM;
    const size_t WB = (size_t)EDIM * EDIM;
    char* ws = (char*)d_ws;
    size_t off = 0;
    auto carve = [&](size_t bytes) {
        char* p = ws + off;
        off += (bytes + 255) & ~(size_t)255;
        return (i8*)p;
    };
    i8* x8  = carve(XB);
    i8* qw8 = carve(WB);
    i8* kw8 = carve(WB);
    i8* vw8 = carve(WB);
    i8* pw8 = carve(WB);
    i8* q8h = carve(XB);
    i8* k8h = carve(XB);
    i8* v8h = carve(XB);
    i8* o8  = carve(XB);

    pack_i8<<<(int)(XB / 4 + 255) / 256, 256, 0, stream>>>(x,   x8,  (int)(XB / 4));
    pack_i8<<<(int)(WB / 4 + 255) / 256, 256, 0, stream>>>(q_w, qw8, (int)(WB / 4));
    pack_i8<<<(int)(WB / 4 + 255) / 256, 256, 0, stream>>>(k_w, kw8, (int)(WB / 4));
    pack_i8<<<(int)(WB / 4 + 255) / 256, 256, 0, stream>>>(v_w, vw8, (int)(WB / 4));
    pack_i8<<<(int)(WB / 4 + 255) / 256, 256, 0, stream>>>(p_w, pw8, (int)(WB / 4));

    dim3 ggrid(MROWS / 64, EDIM / 128);
    qkv_gemm<<<ggrid, 256, 0, stream>>>(x8, qw8, q_b, q8h);
    qkv_gemm<<<ggrid, 256, 0, stream>>>(x8, kw8, k_b, k8h);
    qkv_gemm<<<ggrid, 256, 0, stream>>>(x8, vw8, v_b, v8h);

    attn_head<<<BATCH * NHEADS, 256, SH_TOT, stream>>>(q8h, k8h, v8h, rph, rpw, o8);

    proj_gemm<<<ggrid, 256, 0, stream>>>(o8, pw8, p_b, out);
}
